// GNNModel_38774964748834
// MI455X (gfx1250) — matrix operand order repaired and checked by reading, not compiled, not run on hardware
//
#include <hip/hip_runtime.h>
#include <hip/hip_bf16.h>

// ---------------------------------------------------------------------------
// Types for CDNA5 WMMA (wave32):  D(16x16 f32) = A(16x32 bf16) x B(32x16 bf16) + C
// ---------------------------------------------------------------------------
typedef __attribute__((ext_vector_type(16))) __bf16 v16bf;
typedef __attribute__((ext_vector_type(8)))  float  v8f;

union V16U {
    v16bf v;
    uint4 q[2];
};

__device__ __forceinline__ unsigned short f2bf(float f) {
    unsigned u = __float_as_uint(f);
    unsigned r = u + 0x7FFFu + ((u >> 16) & 1u);   // round-to-nearest-even
    return (unsigned short)(r >> 16);
}

// A fragment: 16x32 bf16.  lane: row = lane&15, half = lane>>4.
//   elems 0..7  <- K = k0 + half*16 + j   (contiguous 8 halves)
//   elems 8..15 <- K = k0 + half*16 + 8 + j   (contiguous 8 halves)
__device__ __forceinline__ v16bf load_a_frag(const unsigned short* p0,
                                             const unsigned short* p1) {
    V16U u;
    u.q[0] = *(const uint4*)p0;
    u.q[1] = *(const uint4*)p1;
    return u.v;
}

// B fragment: 32x16 bf16 from a pre-transposed [N,K] weight.
//   lane: col = n0 + (lane&15); elems j <- K = k0 + (lane>>4)*16 + j
//   => 16 contiguous halves at BT[col*K + k0 + (lane>>4)*16]
__device__ __forceinline__ v16bf load_b_frag(const unsigned short* p) {
    V16U u;
    u.q[0] = ((const uint4*)p)[0];
    u.q[1] = ((const uint4*)p)[1];
    return u.v;
}

__device__ __forceinline__ v8f wmma_bf16(v16bf a, v16bf b, v8f c) {
    return __builtin_amdgcn_wmma_f32_16x16x32_bf16(
        /*neg_a=*/false, a, /*neg_b=*/false, b,
        /*c_mod=*/(short)0, c, /*reuse_a=*/false, /*reuse_b=*/false);
}

#define Hdim 128
#define Gnum 64
#define EDGE_MT 4                       // M-tiles (of 16 edges) per block

// ---------------------------------------------------------------------------
// Small utility kernels
// ---------------------------------------------------------------------------
__global__ void zero_f32_kernel(float* __restrict__ p, int n) {
    int i = blockIdx.x * blockDim.x + threadIdx.x;
    if (i < n) p[i] = 0.0f;
}

__global__ void f32_to_bf16_kernel(const float* __restrict__ s,
                                   unsigned short* __restrict__ d, int n) {
    int i = blockIdx.x * blockDim.x + threadIdx.x;
    if (i < n) d[i] = f2bf(s[i]);
}

// src: [K,Nc] f32 row-major  ->  dst: [Nc,K] bf16 row-major (transposed)
__global__ void transpose_bf16_kernel(const float* __restrict__ src,
                                      unsigned short* __restrict__ dst,
                                      int K, int Nc) {
    int i = blockIdx.x * blockDim.x + threadIdx.x;
    if (i >= K * Nc) return;
    int k = i / Nc, n = i % Nc;
    dst[n * K + k] = f2bf(src[k * Nc + n]);
}

// ---------------------------------------------------------------------------
// Projection: h = relu(x @ proj_w + b)   x:[N,64] bf16, wT:[128,64] bf16
// ---------------------------------------------------------------------------
__global__ __launch_bounds__(128)
void proj_kernel(const unsigned short* __restrict__ xb,
                 const unsigned short* __restrict__ wT,
                 const float* __restrict__ bias,
                 float* __restrict__ hf, unsigned short* __restrict__ hb,
                 int N) {
    const int tid = threadIdx.x, wave = tid >> 5, lane = tid & 31;
    const int ln = lane & 15, lh = lane >> 4;
    const int m0 = blockIdx.x * 16;
    const int n0 = wave * 32;

    int mr = m0 + ln; if (mr > N - 1) mr = N - 1;
    const unsigned short* arow = xb + (size_t)mr * 64;

    v8f c0 = {}, c1 = {};
    #pragma unroll
    for (int k0 = 0; k0 < 64; k0 += 32) {
        v16bf a  = load_a_frag(arow + k0 + lh * 16, arow + k0 + lh * 16 + 8);
        v16bf b0 = load_b_frag(wT + (size_t)(n0 + ln) * 64 + k0 + lh * 16);
        v16bf b1 = load_b_frag(wT + (size_t)(n0 + 16 + ln) * 64 + k0 + lh * 16);
        c0 = wmma_bf16(a, b0, c0);
        c1 = wmma_bf16(a, b1, c1);
    }
    const float bb0 = bias[n0 + ln], bb1 = bias[n0 + 16 + ln];
    #pragma unroll
    for (int r = 0; r < 8; ++r) {
        int m = m0 + lh * 8 + r;
        if (m < N) {
            float v0 = fmaxf(c0[r] + bb0, 0.0f);
            float v1 = fmaxf(c1[r] + bb1, 0.0f);
            size_t o0 = (size_t)m * Hdim + n0 + ln;
            size_t o1 = (size_t)m * Hdim + n0 + 16 + ln;
            hf[o0] = v0; hb[o0] = f2bf(v0);
            hf[o1] = v1; hb[o1] = f2bf(v1);
        }
    }
}

// ---------------------------------------------------------------------------
// Edge message MLP (the hot kernel), 64 edges x 128 cols per block:
//   m = relu(cat[h[dst],h[src]] @ W1 + b1) @ W2 + b2 ; agg[dst] += m
// 4 waves; each wave owns a 32-col slice and iterates 4 M-tiles so every
// B fragment (weight) load is reused 4x from registers.
// ---------------------------------------------------------------------------
__global__ __launch_bounds__(128)
void edge_msg_kernel(const unsigned short* __restrict__ hb,
                     const int* __restrict__ srcI,
                     const int* __restrict__ dstI,
                     const unsigned short* __restrict__ w1T,
                     const float* __restrict__ b1,
                     const unsigned short* __restrict__ w2T,
                     const float* __restrict__ b2,
                     float* __restrict__ agg, int E) {
    __shared__ unsigned short t[EDGE_MT * 16][Hdim];   // relu(GEMM1), bf16

    const int tid = threadIdx.x, wave = tid >> 5, lane = tid & 31;
    const int ln = lane & 15, lh = lane >> 4;
    const int e0 = blockIdx.x * (EDGE_MT * 16);
    const int n0 = wave * 32;

    // Per-lane A-row pointers for the 4 M-tiles.
    const unsigned short* rowD[EDGE_MT];
    const unsigned short* rowS[EDGE_MT];
    #pragma unroll
    for (int mt = 0; mt < EDGE_MT; ++mt) {
        int e = e0 + mt * 16 + ln; if (e > E - 1) e = E - 1;
        rowD[mt] = hb + (size_t)dstI[e] * Hdim;
        rowS[mt] = hb + (size_t)srcI[e] * Hdim;
    }
    // Src rows are consumed only in the 2nd half of the K loop: prefetch them
    // now (each 256B bf16 row spans 2 cache lines).  -> global_prefetch_b8
    #pragma unroll
    for (int mt = 0; mt < EDGE_MT; ++mt) {
        __builtin_prefetch(rowS[mt], 0, 0);
        __builtin_prefetch(rowS[mt] + 64, 0, 0);
    }

    // ---- GEMM1: [64,256] x [256,128] -----------------------------------
    v8f c[EDGE_MT][2];
    #pragma unroll
    for (int mt = 0; mt < EDGE_MT; ++mt) { c[mt][0] = (v8f){}; c[mt][1] = (v8f){}; }

    for (int k0 = 0; k0 < 256; k0 += 32) {
        v16bf bf0 = load_b_frag(w1T + (size_t)(n0 + ln) * 256 + k0 + lh * 16);
        v16bf bf1 = load_b_frag(w1T + (size_t)(n0 + 16 + ln) * 256 + k0 + lh * 16);
        const int kb = k0 & 127;
        const bool first = (k0 < 128);
        #pragma unroll
        for (int mt = 0; mt < EDGE_MT; ++mt) {
            const unsigned short* rp = first ? rowD[mt] : rowS[mt];
            v16bf a = load_a_frag(rp + kb + lh * 16, rp + kb + lh * 16 + 8);
            c[mt][0] = wmma_bf16(a, bf0, c[mt][0]);
            c[mt][1] = wmma_bf16(a, bf1, c[mt][1]);
        }
    }
    const float bb0 = b1[n0 + ln], bb1 = b1[n0 + 16 + ln];
    #pragma unroll
    for (int mt = 0; mt < EDGE_MT; ++mt) {
        #pragma unroll
        for (int r = 0; r < 8; ++r) {
            t[mt * 16 + lh * 8 + r][n0 + ln]      = f2bf(fmaxf(c[mt][0][r] + bb0, 0.0f));
            t[mt * 16 + lh * 8 + r][n0 + 16 + ln] = f2bf(fmaxf(c[mt][1][r] + bb1, 0.0f));
        }
    }
    __syncthreads();

    // ---- GEMM2: [64,128] x [128,128], A from LDS -----------------------
    v8f d[EDGE_MT][2];
    #pragma unroll
    for (int mt = 0; mt < EDGE_MT; ++mt) { d[mt][0] = (v8f){}; d[mt][1] = (v8f){}; }

    for (int k0 = 0; k0 < 128; k0 += 32) {
        v16bf bf0 = load_b_frag(w2T + (size_t)(n0 + ln) * 128 + k0 + lh * 16);
        v16bf bf1 = load_b_frag(w2T + (size_t)(n0 + 16 + ln) * 128 + k0 + lh * 16);
        #pragma unroll
        for (int mt = 0; mt < EDGE_MT; ++mt) {
            v16bf a = load_a_frag(&t[mt * 16 + ln][k0 + lh * 16],
                                  &t[mt * 16 + ln][k0 + lh * 16 + 8]);
            d[mt][0] = wmma_bf16(a, bf0, d[mt][0]);
            d[mt][1] = wmma_bf16(a, bf1, d[mt][1]);
        }
    }
    const float cb0 = b2[n0 + ln], cb1 = b2[n0 + 16 + ln];
    #pragma unroll
    for (int mt = 0; mt < EDGE_MT; ++mt) {
        #pragma unroll
        for (int r = 0; r < 8; ++r) {
            int e = e0 + mt * 16 + lh * 8 + r;
            if (e < E) {
                size_t nb = (size_t)dstI[e] * Hdim;
                atomicAdd(&agg[nb + n0 + ln],      d[mt][0][r] + cb0);
                atomicAdd(&agg[nb + n0 + 16 + ln], d[mt][1][r] + cb1);
            }
        }
    }
}

// ---------------------------------------------------------------------------
// Update + LayerNorm:
//   u = relu(cat[h, agg] @ upd_w + b) ; h = LN(u)*g + b   (in place on h)
// ---------------------------------------------------------------------------
__global__ __launch_bounds__(128)
void update_ln_kernel(unsigned short* __restrict__ hb,
                      const unsigned short* __restrict__ aggb,
                      const unsigned short* __restrict__ wT,   // [128,256]
                      const float* __restrict__ bias,
                      const float* __restrict__ lng,
                      const float* __restrict__ lnb,
                      float* __restrict__ hf, int N) {
    __shared__ float u[16][Hdim];

    const int tid = threadIdx.x, wave = tid >> 5, lane = tid & 31;
    const int ln = lane & 15, lh = lane >> 4;
    const int m0 = blockIdx.x * 16;
    const int n0 = wave * 32;

    int mr = m0 + ln; if (mr > N - 1) mr = N - 1;
    const unsigned short* rowh = hb   + (size_t)mr * Hdim;
    const unsigned short* rowa = aggb + (size_t)mr * Hdim;

    v8f c0 = {}, c1 = {};
    #pragma unroll
    for (int k0 = 0; k0 < 256; k0 += 32) {
        const unsigned short* rp = (k0 < 128) ? rowh : rowa;
        const int kb = k0 & 127;
        v16bf a  = load_a_frag(rp + kb + lh * 16, rp + kb + lh * 16 + 8);
        v16bf b0 = load_b_frag(wT + (size_t)(n0 + ln) * 256 + k0 + lh * 16);
        v16bf b1 = load_b_frag(wT + (size_t)(n0 + 16 + ln) * 256 + k0 + lh * 16);
        c0 = wmma_bf16(a, b0, c0);
        c1 = wmma_bf16(a, b1, c1);
    }
    const float bb0 = bias[n0 + ln], bb1 = bias[n0 + 16 + ln];
    #pragma unroll
    for (int r = 0; r < 8; ++r) {
        u[lh * 8 + r][n0 + ln]      = fmaxf(c0[r] + bb0, 0.0f);
        u[lh * 8 + r][n0 + 16 + ln] = fmaxf(c1[r] + bb1, 0.0f);
    }
    __syncthreads();

    if (tid < 16) {
        int m = m0 + tid;
        if (m < N) {
            float mu = 0.0f;
            #pragma unroll 4
            for (int c = 0; c < Hdim; ++c) mu += u[tid][c];
            mu *= (1.0f / Hdim);
            float var = 0.0f;
            #pragma unroll 4
            for (int c = 0; c < Hdim; ++c) {
                float dd = u[tid][c] - mu;
                var += dd * dd;
            }
            var *= (1.0f / Hdim);
            float rs = rsqrtf(var + 1e-5f);
            #pragma unroll 4
            for (int c = 0; c < Hdim; ++c) {
                float v = (u[tid][c] - mu) * rs * lng[c] + lnb[c];
                size_t o = (size_t)m * Hdim + c;
                hf[o] = v;
                hb[o] = f2bf(v);
            }
        }
    }
}

// ---------------------------------------------------------------------------
// Gate MLP: gate[n] = relu(h@W1+b1)@W2 + b2   (one block of 64 per node)
// ---------------------------------------------------------------------------
__global__ __launch_bounds__(64)
void gate_kernel(const float* __restrict__ hf,
                 const float* __restrict__ w1, const float* __restrict__ b1,
                 const float* __restrict__ w2, const float* __restrict__ b2,
                 float* __restrict__ gate) {
    __shared__ float red[64];
    const int n = blockIdx.x, j = threadIdx.x;
    const float* hr = hf + (size_t)n * Hdim;
    float s = b1[j];
    #pragma unroll 4
    for (int k = 0; k < Hdim; ++k) s += hr[k] * w1[k * 64 + j];
    red[j] = fmaxf(s, 0.0f) * w2[j];
    __syncthreads();
    for (int st = 32; st > 0; st >>= 1) {
        if (j < st) red[j] += red[j + st];
        __syncthreads();
    }
    if (j == 0) gate[n] = red[0] + b2[0];
}

// ---------------------------------------------------------------------------
// Segment bookkeeping (batch is sorted)
// ---------------------------------------------------------------------------
__global__ void seg_init_kernel(int* gstart, int* gend) {
    int g = threadIdx.x;
    if (g < Gnum) { gstart[g] = 0; gend[g] = 0; }
}

__global__ void seg_bounds_kernel(const int* __restrict__ batch,
                                  int* __restrict__ gstart,
                                  int* __restrict__ gend, int N) {
    int n = blockIdx.x * blockDim.x + threadIdx.x;
    if (n >= N) return;
    int g = batch[n];
    if (n == 0 || batch[n - 1] != g) gstart[g] = n;
    if (n == N - 1 || batch[n + 1] != g) gend[g] = n + 1;
}

__global__ __launch_bounds__(128)
void group_stats_kernel(const float* __restrict__ gate,
                        const int* __restrict__ gstart,
                        const int* __restrict__ gend,
                        float* __restrict__ gmax, float* __restrict__ denom) {
    __shared__ float red[128];
    const int g = blockIdx.x, tid = threadIdx.x;
    const int s = gstart[g], e = gend[g];
    float mx = -3.4e38f;
    for (int n = s + tid; n < e; n += 128) mx = fmaxf(mx, gate[n]);
    red[tid] = mx; __syncthreads();
    for (int st = 64; st > 0; st >>= 1) {
        if (tid < st) red[tid] = fmaxf(red[tid], red[tid + st]);
        __syncthreads();
    }
    const float m = red[0]; __syncthreads();
    float sm = 0.0f;
    for (int n = s + tid; n < e; n += 128) sm += __expf(gate[n] - m);
    red[tid] = sm; __syncthreads();
    for (int st = 64; st > 0; st >>= 1) {
        if (tid < st) red[tid] += red[tid + st];
        __syncthreads();
    }
    if (tid == 0) { gmax[g] = m; denom[g] = red[0]; }
}

__global__ __launch_bounds__(128)
void pooled_kernel(const float* __restrict__ hf, const float* __restrict__ gate,
                   const int* __restrict__ gstart, const int* __restrict__ gend,
                   const float* __restrict__ gmax, const float* __restrict__ denom,
                   float* __restrict__ pooled) {
    const int g = blockIdx.x, c = threadIdx.x;
    const int s = gstart[g], e = gend[g];
    const float dm = denom[g];
    const float inv = (dm > 0.0f) ? 1.0f / dm : 0.0f;
    const float m = gmax[g];
    float acc = 0.0f;
    for (int n = s; n < e; ++n) {
        float w = __expf(gate[n] - m) * inv;
        acc += hf[(size_t)n * Hdim + c] * w;
    }
    pooled[g * Hdim + c] = acc;
}

__global__ __launch_bounds__(128)
void head_kernel(const float* __restrict__ pooled,
                 const float* __restrict__ fc1w, const float* __restrict__ fc1b,
                 const float* __restrict__ fc2w, const float* __restrict__ fc2b,
                 float* __restrict__ out) {
    __shared__ float red[128];
    const int g = blockIdx.x, j = threadIdx.x;
    const float* pr = pooled + g * Hdim;
    float s = fc1b[j];
    #pragma unroll 4
    for (int k = 0; k < Hdim; ++k) s += pr[k] * fc1w[k * Hdim + j];
    red[j] = fmaxf(s, 0.0f) * fc2w[j];
    __syncthreads();
    for (int st = 64; st > 0; st >>= 1) {
        if (j < st) red[j] += red[j + st];
        __syncthreads();
    }
    if (j == 0) out[g] = red[0] + fc2b[0];
}

// ---------------------------------------------------------------------------
// Host orchestration
// ---------------------------------------------------------------------------
extern "C" void kernel_launch(void* const* d_in, const int* in_sizes, int n_in,
                              void* d_out, int out_size, void* d_ws, size_t ws_size,
                              hipStream_t stream) {
    const int D_IN = 64;
    const int N = in_sizes[0] / D_IN;     // 20000
    const int E = in_sizes[1] / 2;        // 640000

    const float* x        = (const float*)d_in[0];
    const int*   ei       = (const int*)d_in[1];   // [2,E]: src row then dst row
    const int*   batch    = (const int*)d_in[2];
    const float* proj_w   = (const float*)d_in[3];
    const float* proj_b   = (const float*)d_in[4];
    const float* msg_w1   = (const float*)d_in[5];
    const float* msg_b1   = (const float*)d_in[6];
    const float* msg_w2   = (const float*)d_in[7];
    const float* msg_b2   = (const float*)d_in[8];
    const float* upd_w    = (const float*)d_in[9];
    const float* upd_b    = (const float*)d_in[10];
    const float* ln_g     = (const float*)d_in[11];
    const float* ln_b     = (const float*)d_in[12];
    const float* gate_w1  = (const float*)d_in[13];
    const float* gate_b1  = (const float*)d_in[14];
    const float* gate_w2  = (const float*)d_in[15];
    const float* gate_b2  = (const float*)d_in[16];
    const float* fc1_w    = (const float*)d_in[17];
    const float* fc1_b    = (const float*)d_in[18];
    const float* fc2_w    = (const float*)d_in[19];
    const float* fc2_b    = (const float*)d_in[20];
    float* out = (float*)d_out;

    // ---- workspace layout (256B aligned) -------------------------------
    char* base = (char*)d_ws;
    size_t off = 0;
    auto take = [&](size_t bytes) -> char* {
        char* p = base + off;
        off = (off + bytes + 255) & ~(size_t)255;
        return p;
    };
    float*          h_f32   = (float*)         take((size_t)N * Hdim * 4);
    unsigned short* h_bf16  = (unsigned short*)take((size_t)N * Hdim * 2);
    float*          agg_f32 = (float*)         take((size_t)N * Hdim * 4);
    unsigned short* agg_b16 = (unsigned short*)take((size_t)N * Hdim * 2);
    unsigned short* x_b16   = (unsigned short*)take((size_t)N * D_IN * 2);
    unsigned short* projT   = (unsigned short*)take((size_t)Hdim * D_IN * 2);
    unsigned short* w1T     = (unsigned short*)take((size_t)4 * Hdim * 256 * 2);
    unsigned short* w2T     = (unsigned short*)take((size_t)4 * Hdim * Hdim * 2);
    unsigned short* updT    = (unsigned short*)take((size_t)4 * Hdim * 256 * 2);
    float*          gate    = (float*)         take((size_t)N * 4);
    float*          gmax    = (float*)         take(Gnum * 4);
    float*          denom   = (float*)         take(Gnum * 4);
    int*            gstart  = (int*)           take(Gnum * 4);
    int*            gend    = (int*)           take(Gnum * 4);
    float*          pooled  = (float*)         take((size_t)Gnum * Hdim * 4);
    (void)ws_size;

    const int nodeTiles = (N + 15) / 16;
    const int edgeTiles = (E + EDGE_MT * 16 - 1) / (EDGE_MT * 16);

    // ---- weight prep ---------------------------------------------------
    {
        int n = D_IN * Hdim;
        transpose_bf16_kernel<<<(n + 255) / 256, 256, 0, stream>>>(proj_w, projT, D_IN, Hdim);
    }
    for (int i = 0; i < 4; ++i) {
        int n1 = 256 * Hdim, n2 = Hdim * Hdim;
        transpose_bf16_kernel<<<(n1 + 255) / 256, 256, 0, stream>>>(
            msg_w1 + (size_t)i * n1, w1T + (size_t)i * n1, 256, Hdim);
        transpose_bf16_kernel<<<(n2 + 255) / 256, 256, 0, stream>>>(
            msg_w2 + (size_t)i * n2, w2T + (size_t)i * n2, Hdim, Hdim);
        transpose_bf16_kernel<<<(n1 + 255) / 256, 256, 0, stream>>>(
            upd_w + (size_t)i * n1, updT + (size_t)i * n1, 256, Hdim);
    }
    {
        int n = N * D_IN;
        f32_to_bf16_kernel<<<(n + 255) / 256, 256, 0, stream>>>(x, x_b16, n);
    }

    // ---- projection ----------------------------------------------------
    proj_kernel<<<nodeTiles, 128, 0, stream>>>(x_b16, projT, proj_b, h_f32, h_bf16, N);

    // ---- 4 message-passing layers --------------------------------------
    const int* srcI = ei;
    const int* dstI = ei + E;
    for (int i = 0; i < 4; ++i) {
        int nh = N * Hdim;
        zero_f32_kernel<<<(nh + 255) / 256, 256, 0, stream>>>(agg_f32, nh);
        edge_msg_kernel<<<edgeTiles, 128, 0, stream>>>(
            h_bf16, srcI, dstI,
            w1T + (size_t)i * 256 * Hdim, msg_b1 + i * Hdim,
            w2T + (size_t)i * Hdim * Hdim, msg_b2 + i * Hdim,
            agg_f32, E);
        f32_to_bf16_kernel<<<(nh + 255) / 256, 256, 0, stream>>>(agg_f32, agg_b16, nh);
        update_ln_kernel<<<nodeTiles, 128, 0, stream>>>(
            h_bf16, agg_b16,
            updT + (size_t)i * 256 * Hdim, upd_b + i * Hdim,
            ln_g + i * Hdim, ln_b + i * Hdim,
            h_f32, N);
    }

    // ---- attention pooling + head --------------------------------------
    gate_kernel<<<N, 64, 0, stream>>>(h_f32, gate_w1, gate_b1, gate_w2, gate_b2, gate);
    seg_init_kernel<<<1, Gnum, 0, stream>>>(gstart, gend);
    seg_bounds_kernel<<<(N + 255) / 256, 256, 0, stream>>>(batch, gstart, gend, N);
    group_stats_kernel<<<Gnum, 128, 0, stream>>>(gate, gstart, gend, gmax, denom);
    pooled_kernel<<<Gnum, 128, 0, stream>>>(h_f32, gate, gstart, gend, gmax, denom, pooled);
    head_kernel<<<Gnum, 128, 0, stream>>>(pooled, fc1_w, fc1_b, fc2_w, fc2_b, out);
    (void)out_size; (void)n_in; (void)gate_b2;
}
